// GlobalTransformerLayer_70789650972750
// MI455X (gfx1250) — compile-verified, matrix operand order checked by reading, not run on hardware
//
#include <hip/hip_runtime.h>
#include <hip/hip_bf16.h>

typedef __attribute__((ext_vector_type(16))) _Float16 v16h;
typedef __attribute__((ext_vector_type(8)))  float    v8f;

#define DIM_   256
#define HEADS_ 8
#define HD_    32
#define B_     16
#define N_     512
#define ROWS_  (B_ * N_)      // 8192
#define MAXD_  128
#define NG_    32

#if defined(__gfx1250__) && __has_builtin(__builtin_amdgcn_tensor_load_to_lds) && \
    __has_builtin(__builtin_amdgcn_s_wait_tensorcnt)
#define HAVE_TDM 1
#else
#define HAVE_TDM 0
#endif

union FragU { v16h h; uint4 u[2]; };

// A-style fragment (16x32 f16, MxK): lane 0-15 -> row M=lane, K=0..7 & 16..23;
// lane 16-31 -> row M=lane-16, K=8..15 & 24..31. Two aligned 16B loads.
__device__ __forceinline__ v16h load_frag_rowk(const _Float16* __restrict__ base, int ld) {
  int lane = threadIdx.x & 31;
  int r    = lane & 15;
  int klo  = (lane >> 4) << 3;          // 0 or 8
  const _Float16* p = base + (size_t)r * ld;
  FragU f;
  f.u[0] = *reinterpret_cast<const uint4*>(p + klo);
  f.u[1] = *reinterpret_cast<const uint4*>(p + 16 + klo);
  return f.h;
}

// B-style fragment from a [K,N]-stored matrix (element B[k][n] = base[k*ld + n]).
__device__ __forceinline__ v16h load_frag_colk(const _Float16* __restrict__ base, int ld) {
  int lane = threadIdx.x & 31;
  int n    = lane & 15;
  int klo  = (lane >> 4) << 3;
  v16h f;
#pragma unroll
  for (int j = 0; j < 8; ++j) f[j]     = base[(size_t)(klo + j) * ld + n];
#pragma unroll
  for (int j = 0; j < 8; ++j) f[8 + j] = base[(size_t)(16 + klo + j) * ld + n];
  return f;
}

__device__ __forceinline__ v8f wmma_f16f32(v16h a, v16h b, v8f c) {
  return __builtin_amdgcn_wmma_f32_16x16x32_f16(false, a, false, b, (short)0, c, false, false);
}

__device__ __forceinline__ float gelu_exact(float x) {
  return 0.5f * x * (1.0f + erff(x * 0.70710678118654752f));
}

#if HAVE_TDM
typedef __attribute__((ext_vector_type(4))) unsigned tdm_v4u;
typedef __attribute__((ext_vector_type(8))) int      tdm_v8i;
typedef __attribute__((ext_vector_type(4))) int      tdm_v4i;

// DMA a 2D f16 tile [tile_h rows x tile_w elems] from global (row stride in
// elements) into LDS (rows packed contiguously). Tracked on TENSORcnt.
__device__ __forceinline__ void tdm_load_2d(const _Float16* lds_dst,
                                            const _Float16* gsrc,
                                            int tile_w, int tile_h,
                                            unsigned row_stride_elems) {
  unsigned lds_off = (unsigned)(size_t)lds_dst;   // LDS aperture: low 32 bits = LDS addr
  unsigned long long ga = (unsigned long long)(size_t)gsrc;
  tdm_v4u g0 = { 1u,                                    // count=1 (valid user D#)
                 lds_off,                               // lds_addr
                 (unsigned)(ga & 0xFFFFFFFFu),          // global_addr[31:0]
                 (unsigned)((ga >> 32) & 0x01FFFFFFu) | (2u << 30) }; // addr[56:32] | type=2
  tdm_v8i g1 = { (int)(1u << 16),                       // workgroup_mask=0, data_size=2B
                 (int)0xFFFF0000u,                      // tensor_dim0 = 0xFFFF (low16)
                 (int)0xFFFF0000u,                      // dim0 hi=0, tensor_dim1 low16=0xFFFF
                 (int)((unsigned)tile_w << 16),         // tensor_dim1 hi=0, tile_dim0
                 tile_h,                                // tile_dim1 (tile_dim2 = 0)
                 (int)row_stride_elems,                 // tensor_dim0_stride[31:0]
                 0, 0 };
  tdm_v4i gz = { 0, 0, 0, 0 };
#if __clang_major__ >= 23
  tdm_v8i gz8 = { 0, 0, 0, 0, 0, 0, 0, 0 };
  __builtin_amdgcn_tensor_load_to_lds(g0, g1, gz, gz, gz8, 0);
#else
  __builtin_amdgcn_tensor_load_to_lds(g0, g1, gz, gz, 0);
#endif
}
#endif

// ---------------------------------------------------------------- utilities
__global__ void __launch_bounds__(256) k_f32_to_f16(const float* __restrict__ src,
                                                    _Float16* __restrict__ dst, int n) {
  int i = blockIdx.x * 256 + threadIdx.x;
  if (i < n) dst[i] = (_Float16)src[i];
}

// one wave per row of 256
__global__ void __launch_bounds__(256) k_layernorm(const float* __restrict__ x,
                                                   const float* __restrict__ g,
                                                   const float* __restrict__ bta,
                                                   float* __restrict__ outF32,
                                                   _Float16* __restrict__ outF16) {
  int wave = threadIdx.x >> 5, lane = threadIdx.x & 31;
  int row  = blockIdx.x * 8 + wave;
  const float* xr = x + (size_t)row * DIM_;
  float v[8], s = 0.f, sq = 0.f;
#pragma unroll
  for (int i = 0; i < 8; ++i) { v[i] = xr[lane + i * 32]; s += v[i]; sq += v[i] * v[i]; }
#pragma unroll
  for (int m = 1; m < 32; m <<= 1) { s += __shfl_xor(s, m, 32); sq += __shfl_xor(sq, m, 32); }
  float mean = s * (1.0f / DIM_);
  float var  = sq * (1.0f / DIM_) - mean * mean;
  float rs   = rsqrtf(var + 1e-5f);
#pragma unroll
  for (int i = 0; i < 8; ++i) {
    int c = lane + i * 32;
    float o = (v[i] - mean) * rs * g[c] + bta[c];
    if (outF32) outF32[(size_t)row * DIM_ + c] = o;
    if (outF16) outF16[(size_t)row * DIM_ + c] = (_Float16)o;
  }
}

// ------------------------------------------------- attention bias [B,H,N,N] f16
__global__ void __launch_bounds__(256) k_bias(const int* __restrict__ dist,
                                              const float* __restrict__ d3,
                                              const float* __restrict__ emb,    // [128,8]
                                              const float* __restrict__ rbfw,   // [8,32]
                                              const float* __restrict__ rbfb,   // [8]
                                              _Float16* __restrict__ bias) {
  __shared__ float s_emb[MAXD_ * HEADS_];
  __shared__ float s_w[HEADS_ * NG_];
  __shared__ float s_b[HEADS_];
  int t = threadIdx.x;
#pragma unroll
  for (int i = 0; i < 4; ++i) s_emb[t + i * 256] = emb[t + i * 256];
  s_w[t % (HEADS_ * NG_)] = rbfw[t % (HEADS_ * NG_)];
  if (t < HEADS_) s_b[t] = rbfb[t];
  __syncthreads();

  size_t idx = (size_t)blockIdx.x * 256 + t;         // b*N*N + q*N + k
  int k  = (int)(idx & 511);
  int q  = (int)((idx >> 9) & 511);
  int bb = (int)(idx >> 18);
  int d  = dist[idx]; d = d < 0 ? 0 : (d > MAXD_ - 1 ? MAXD_ - 1 : d);
  float dv = d3[idx];

  const float coeff = -1.20125f;                      // -0.5/((20/31)^2)
  float r[NG_];
#pragma unroll
  for (int g = 0; g < NG_; ++g) {
    float diff = dv - (float)g * 0.6451612903225806f; // linspace(0,20,32)
    r[g] = __expf(coeff * diff * diff);
  }
#pragma unroll
  for (int h = 0; h < HEADS_; ++h) {
    float a = s_b[h] + s_emb[d * HEADS_ + h];
#pragma unroll
    for (int g = 0; g < NG_; ++g) a += r[g] * s_w[h * NG_ + g];
    bias[(((size_t)bb * HEADS_ + h) * N_ + q) * N_ + k] = (_Float16)a;
  }
}

// ----------------------------------------------------- WMMA GEMM: C = A @ W^T
// A: [M,K] f16 row-major, W: [N,K] f16 row-major; fused bias / GELU / residual.
// Block tile 128x64, wave tile 32x32 (2x2 fragments) -- this register budget is
// what lets the scheduler software-pipeline loads against WMMAs (round-1 codegen).
__global__ void __launch_bounds__(256) k_gemm(const _Float16* __restrict__ A, int lda,
                                              const _Float16* __restrict__ W, int ldw,
                                              const float* __restrict__ bias,
                                              const float* __restrict__ residual,
                                              float* __restrict__ outF32,
                                              _Float16* __restrict__ outF16,
                                              int N, int K, int gelu) {
  int wave = threadIdx.x >> 5, lane = threadIdx.x & 31;
  int wm = wave >> 1, wn = wave & 1;
  int row0 = blockIdx.y * 128 + wm * 32;
  int col0 = blockIdx.x * 64 + wn * 32;

  v8f acc[2][2] = {{{0.f,0.f,0.f,0.f,0.f,0.f,0.f,0.f},{0.f,0.f,0.f,0.f,0.f,0.f,0.f,0.f}},
                   {{0.f,0.f,0.f,0.f,0.f,0.f,0.f,0.f},{0.f,0.f,0.f,0.f,0.f,0.f,0.f,0.f}}};
  for (int k = 0; k < K; k += 32) {
    v16h a0 = load_frag_rowk(A + (size_t)row0 * lda + k, lda);
    v16h a1 = load_frag_rowk(A + (size_t)(row0 + 16) * lda + k, lda);
    v16h b0 = load_frag_rowk(W + (size_t)col0 * ldw + k, ldw);
    v16h b1 = load_frag_rowk(W + (size_t)(col0 + 16) * ldw + k, ldw);
    acc[0][0] = wmma_f16f32(a0, b0, acc[0][0]);
    acc[0][1] = wmma_f16f32(a0, b1, acc[0][1]);
    acc[1][0] = wmma_f16f32(a1, b0, acc[1][0]);
    acc[1][1] = wmma_f16f32(a1, b1, acc[1][1]);
  }
  int nl = lane & 15;
  int hsel = (lane >> 4) << 3;
#pragma unroll
  for (int i = 0; i < 2; ++i)
#pragma unroll
    for (int j = 0; j < 2; ++j)
#pragma unroll
      for (int r = 0; r < 8; ++r) {
        int row = row0 + i * 16 + r + hsel;
        int col = col0 + j * 16 + nl;
        float v = acc[i][j][r] + bias[col];
        if (gelu) v = gelu_exact(v);
        if (residual) v += residual[(size_t)row * N + col];
        if (outF32) outF32[(size_t)row * N + col] = v;
        if (outF16) outF16[(size_t)row * N + col] = (_Float16)v;
      }
}

// -------------------------------------------- flash attention, wave per q-tile
// Bias (16x32) and V (32x32) tiles are DMA'd into LDS by the Tensor Data Mover,
// double-buffered so the next tile's DMA overlaps this tile's WMMA work.
__global__ void __launch_bounds__(256) k_attn(const _Float16* __restrict__ qkv, // [8192,768]
                                              const _Float16* __restrict__ bias,// [B,H,N,N]
                                              _Float16* __restrict__ ctx) {     // [8192,256]
  __shared__ __align__(16) _Float16 pbuf[8][16 * 32];
#if HAVE_TDM
  __shared__ __align__(16) _Float16 bbuf[8][2][16 * 32];
  __shared__ __align__(16) _Float16 vbuf[8][2][32 * 32];
#endif
  int wave = threadIdx.x >> 5, lane = threadIdx.x & 31;
  int gw = blockIdx.x * 8 + wave;
  int qt = gw & 31;            // q tile (16 rows each)
  int bh = gw >> 5;
  int h  = bh & 7;
  int b  = bh >> 3;
  const _Float16* Q  = qkv + ((size_t)(b * N_ + qt * 16)) * 768 + h * HD_;
  const _Float16* Kp = qkv + (size_t)(b * N_) * 768 + DIM_ + h * HD_;
  const _Float16* Vp = qkv + (size_t)(b * N_) * 768 + 2 * DIM_ + h * HD_;
  const _Float16* Bp = bias + (((size_t)b * HEADS_ + h) * N_ + qt * 16) * N_;

  v16h qf = load_frag_rowk(Q, 768);   // 16 q-rows x 32 hd
  float mrow[8], lrow[8];
  v8f o0 = {0.f,0.f,0.f,0.f,0.f,0.f,0.f,0.f};
  v8f o1 = {0.f,0.f,0.f,0.f,0.f,0.f,0.f,0.f};
#pragma unroll
  for (int r = 0; r < 8; ++r) { mrow[r] = -1e30f; lrow[r] = 0.f; }
  const float scale = 0.17677669529663687f;   // 1/sqrt(32)
  const v8f zacc = {0.f,0.f,0.f,0.f,0.f,0.f,0.f,0.f};
  int nl = lane & 15;
  int hsel = (lane >> 4) << 3;

#if HAVE_TDM
  // prologue: DMA tile kb=0 into buffer 0
  tdm_load_2d(&bbuf[wave][0][0], Bp, 32, 16, N_);
  tdm_load_2d(&vbuf[wave][0][0], Vp, 32, 32, 768);
#endif

  for (int kb = 0; kb < N_; kb += 32) {
    v16h kf0 = load_frag_rowk(Kp + (size_t)kb * 768, 768);
    v16h kf1 = load_frag_rowk(Kp + (size_t)(kb + 16) * 768, 768);

    float bv0[8], bv1[8];
    v16h v0f, v1f;
#if HAVE_TDM
    int bufc = (kb >> 5) & 1;
    __builtin_amdgcn_s_wait_tensorcnt(0);
    {
      const _Float16* bb = &bbuf[wave][bufc][0];
#pragma unroll
      for (int r = 0; r < 8; ++r) {
        int row = r + hsel;
        bv0[r] = (float)bb[row * 32 + nl];
        bv1[r] = (float)bb[row * 32 + 16 + nl];
      }
      v0f = load_frag_colk(&vbuf[wave][bufc][0], 32);
      v1f = load_frag_colk(&vbuf[wave][bufc][0] + 16, 32);
    }
    if (kb + 32 < N_) {  // kick off next tile's DMA; overlaps compute below
      tdm_load_2d(&bbuf[wave][1 - bufc][0], Bp + (kb + 32), 32, 16, N_);
      tdm_load_2d(&vbuf[wave][1 - bufc][0], Vp + (size_t)(kb + 32) * 768, 32, 32, 768);
    }
#else
#pragma unroll
    for (int r = 0; r < 8; ++r) {
      int row = r + hsel;
      bv0[r] = (float)Bp[(size_t)row * N_ + kb + nl];
      bv1[r] = (float)Bp[(size_t)row * N_ + kb + 16 + nl];
    }
    v0f = load_frag_colk(Vp + (size_t)kb * 768, 768);
    v1f = load_frag_colk(Vp + (size_t)kb * 768 + 16, 768);
#endif

    v8f s0 = wmma_f16f32(qf, kf0, zacc);
    v8f s1 = wmma_f16f32(qf, kf1, zacc);
#pragma unroll
    for (int r = 0; r < 8; ++r) {
      s0[r] = s0[r] * scale + bv0[r];
      s1[r] = s1[r] * scale + bv1[r];
    }
    // online softmax per row (row's 16 cols live in one 16-lane half)
#pragma unroll
    for (int r = 0; r < 8; ++r) {
      float mx = fmaxf(s0[r], s1[r]);
#pragma unroll
      for (int m = 1; m < 16; m <<= 1) mx = fmaxf(mx, __shfl_xor(mx, m, 32));
      float nm = fmaxf(mrow[r], mx);
      float sc = __expf(mrow[r] - nm);
      mrow[r] = nm;
      float p0 = __expf(s0[r] - nm);
      float p1 = __expf(s1[r] - nm);
      s0[r] = p0; s1[r] = p1;
      float rs = p0 + p1;
#pragma unroll
      for (int m = 1; m < 16; m <<= 1) rs += __shfl_xor(rs, m, 32);
      lrow[r] = lrow[r] * sc + rs;
      o0[r] *= sc;
      o1[r] *= sc;
    }
    // D-layout -> A-layout via LDS
    __syncthreads();
    _Float16* pb = pbuf[wave];
#pragma unroll
    for (int r = 0; r < 8; ++r) {
      int row = r + hsel;
      pb[row * 32 + nl]      = (_Float16)s0[r];
      pb[row * 32 + 16 + nl] = (_Float16)s1[r];
    }
    __syncthreads();
    v16h pf = load_frag_rowk(pb, 32);
    o0 = wmma_f16f32(pf, v0f, o0);
    o1 = wmma_f16f32(pf, v1f, o1);
  }
#pragma unroll
  for (int r = 0; r < 8; ++r) {
    int row = qt * 16 + r + hsel;
    float inv = 1.0f / lrow[r];
    size_t base = ((size_t)(b * N_ + row)) * DIM_ + h * HD_;
    ctx[base + nl]      = (_Float16)(o0[r] * inv);
    ctx[base + 16 + nl] = (_Float16)(o1[r] * inv);
  }
}

// ---------------------------------------------------------------------------
extern "C" void kernel_launch(void* const* d_in, const int* in_sizes, int n_in,
                              void* d_out, int out_size, void* d_ws, size_t ws_size,
                              hipStream_t stream) {
  (void)in_sizes; (void)n_in; (void)out_size; (void)ws_size;
  const float* x        = (const float*)d_in[0];
  const int*   dist     = (const int*)  d_in[2];
  const float* d3       = (const float*)d_in[3];
  const float* n1g      = (const float*)d_in[4];
  const float* n1b      = (const float*)d_in[5];
  const float* inw      = (const float*)d_in[6];
  const float* inb      = (const float*)d_in[7];
  const float* outw     = (const float*)d_in[8];
  const float* outb     = (const float*)d_in[9];
  const float* emb      = (const float*)d_in[10];
  const float* rbfw     = (const float*)d_in[11];
  const float* rbfb     = (const float*)d_in[12];
  const float* n2g      = (const float*)d_in[13];
  const float* n2b      = (const float*)d_in[14];
  const float* w1       = (const float*)d_in[15];
  const float* b1       = (const float*)d_in[16];
  const float* w2       = (const float*)d_in[17];
  const float* b2       = (const float*)d_in[18];
  float* out            = (float*)d_out;

  char* cur = (char*)d_ws;
  auto alloc = [&](size_t bytes) -> void* {
    void* p = (void*)cur;
    cur += (bytes + 255) & ~(size_t)255;
    return p;
  };
  float*    h_f32  = (float*)   alloc((size_t)ROWS_ * DIM_ * 4);
  _Float16* h_f16  = (_Float16*)alloc((size_t)ROWS_ * DIM_ * 2);
  _Float16* inw16  = (_Float16*)alloc((size_t)768 * 256 * 2);
  _Float16* outw16 = (_Float16*)alloc((size_t)256 * 256 * 2);
  _Float16* w116   = (_Float16*)alloc((size_t)1024 * 256 * 2);
  _Float16* w216   = (_Float16*)alloc((size_t)256 * 1024 * 2);
  _Float16* qkv16  = (_Float16*)alloc((size_t)ROWS_ * 768 * 2);
  _Float16* bias16 = (_Float16*)alloc((size_t)B_ * HEADS_ * N_ * N_ * 2);
  _Float16* ctx16  = (_Float16*)alloc((size_t)ROWS_ * DIM_ * 2);
  float*    hattn  = (float*)   alloc((size_t)ROWS_ * DIM_ * 4);
  _Float16* h216   = (_Float16*)alloc((size_t)ROWS_ * DIM_ * 2);
  _Float16* ff116  = (_Float16*)alloc((size_t)ROWS_ * 1024 * 2);

  // weight conversion fp32 -> fp16
  k_f32_to_f16<<<(768 * 256 + 255) / 256, 256, 0, stream>>>(inw, inw16, 768 * 256);
  k_f32_to_f16<<<(256 * 256 + 255) / 256, 256, 0, stream>>>(outw, outw16, 256 * 256);
  k_f32_to_f16<<<(1024 * 256 + 255) / 256, 256, 0, stream>>>(w1, w116, 1024 * 256);
  k_f32_to_f16<<<(256 * 1024 + 255) / 256, 256, 0, stream>>>(w2, w216, 256 * 1024);

  // h = layernorm1(x)
  k_layernorm<<<ROWS_ / 8, 256, 0, stream>>>(x, n1g, n1b, h_f32, h_f16);

  // attention bias [B,H,N,N] f16
  k_bias<<<(B_ * N_ * N_) / 256, 256, 0, stream>>>(dist, d3, emb, rbfw, rbfb, bias16);

  // qkv = h @ in_proj_w^T + in_proj_b  -> f16 [8192,768]
  k_gemm<<<dim3(768 / 64, ROWS_ / 128), 256, 0, stream>>>(
      h_f16, 256, inw16, 256, inb, nullptr, nullptr, qkv16, 768, 256, 0);

  // flash attention with fused bias (TDM-staged tiles) -> ctx f16 [8192,256]
  k_attn<<<(B_ * HEADS_ * (N_ / 16)) / 8, 256, 0, stream>>>(qkv16, bias16, ctx16);

  // h_attn = h + ctx @ out_proj_w^T + out_proj_b
  k_gemm<<<dim3(256 / 64, ROWS_ / 128), 256, 0, stream>>>(
      ctx16, 256, outw16, 256, outb, h_f32, hattn, nullptr, 256, 256, 0);

  // h2 = layernorm2(h_attn)
  k_layernorm<<<ROWS_ / 8, 256, 0, stream>>>(hattn, n2g, n2b, nullptr, h216);

  // ff1 = gelu(h2 @ ffn_w1^T + b1) -> f16 [8192,1024]
  k_gemm<<<dim3(1024 / 64, ROWS_ / 128), 256, 0, stream>>>(
      h216, 256, w116, 256, b1, nullptr, nullptr, ff116, 1024, 256, 1);

  // out = h_attn + ff1 @ ffn_w2^T + b2
  k_gemm<<<dim3(256 / 64, ROWS_ / 128), 256, 0, stream>>>(
      ff116, 1024, w216, 1024, b2, hattn, out, nullptr, 256, 1024, 0);
}
